// BertModelWrapper_noise_46832323396259
// MI455X (gfx1250) — compile-verified, matrix operand order checked by reading, not run on hardware
//
#include <hip/hip_runtime.h>
#include <math.h>

// ---------------------------------------------------------------------------
// BERT-base forward for MI455X (gfx1250, wave32, WMMA).
// All GEMMs run on v_wmma_f32_16x16x32_bf16 (fp32 accumulate).
// Main GEMM stages the weight tile in LDS via async global->LDS copies
// (ASYNCcnt) with double buffering; B fragments then come from ds_load_b128.
// Fragment loads are hoisted ahead of the WMMA chain so the backend can emit
// one load clause + one wait + a back-to-back WMMA burst.
// ---------------------------------------------------------------------------

typedef __bf16 bf16_t;
typedef __attribute__((ext_vector_type(8)))  __bf16 v8bf;
typedef __attribute__((ext_vector_type(16))) __bf16 v16bf;
typedef __attribute__((ext_vector_type(8)))  float  v8f;
typedef int v4i __attribute__((vector_size(4 * sizeof(int))));

constexpr int CB  = 8;
constexpr int CS  = 256;
constexpr int CH  = 768;
constexpr int CL  = 12;
constexpr int CNH = 12;
constexpr int CDH = 64;
constexpr int CFF = 3072;
constexpr int CM  = CB * CS;        // 2048 tokens
constexpr int QKVN = 3 * CH;        // 2304 fused QKV output dim

// ---------------------------------------------------------------------------
// Async global -> LDS copy of one 16B packet (CDNA5, tracked by ASYNCcnt)
// ---------------------------------------------------------------------------
__device__ __forceinline__ void async_copy_b128(bf16_t* lds_dst,
                                                const bf16_t* gsrc) {
#if __has_builtin(__builtin_amdgcn_global_load_async_to_lds_b128)
  __builtin_amdgcn_global_load_async_to_lds_b128(
      (__attribute__((address_space(1))) v4i*)(uintptr_t)gsrc,
      (__attribute__((address_space(3))) v4i*)(unsigned)(uintptr_t)lds_dst,
      0, 0);
#else
  unsigned loff = (unsigned)(uintptr_t)lds_dst;   // low 32 bits = LDS offset
  asm volatile("global_load_async_to_lds_b128 %0, %1, off"
               :: "v"(loff), "v"(gsrc) : "memory");
#endif
}

__device__ __forceinline__ void wait_async_le(int n) {
#if __has_builtin(__builtin_amdgcn_s_wait_asynccnt)
  if (n == 0) __builtin_amdgcn_s_wait_asynccnt(0);
  else        __builtin_amdgcn_s_wait_asynccnt(2);
#else
  if (n == 0) asm volatile("s_wait_asynccnt 0x0" ::: "memory");
  else        asm volatile("s_wait_asynccnt 0x2" ::: "memory");
#endif
}

// ---------------------------------------------------------------------------
// WMMA fragment loader (16-bit, 16x32 operand, row-major source with stride ld)
// ---------------------------------------------------------------------------
__device__ __forceinline__ v16bf load_frag(const bf16_t* __restrict__ base,
                                           int row0, int ld, int kk,
                                           unsigned lane) {
  const bf16_t* p = base + (size_t)(row0 + (lane & 15)) * ld + kk + ((lane & 16) ? 8 : 0);
  v8bf lo = *reinterpret_cast<const v8bf*>(p);
  v8bf hi = *reinterpret_cast<const v8bf*>(p + 16);
  return __builtin_shufflevector(lo, hi, 0, 1, 2, 3, 4, 5, 6, 7,
                                 8, 9, 10, 11, 12, 13, 14, 15);
}

// B fragment out of the LDS stage buffer (row stride = 64 bf16 = 128B)
__device__ __forceinline__ v16bf lds_frag(const bf16_t* buf, int rowbase,
                                          int kstep, unsigned lane) {
  const bf16_t* p = buf + (size_t)(rowbase + (lane & 15)) * 64 + kstep * 32 +
                    ((lane & 16) ? 8 : 0);
  v8bf lo = *reinterpret_cast<const v8bf*>(p);
  v8bf hi = *reinterpret_cast<const v8bf*>(p + 16);
  return __builtin_shufflevector(lo, hi, 0, 1, 2, 3, 4, 5, 6, 7,
                                 8, 9, 10, 11, 12, 13, 14, 15);
}

// ---------------------------------------------------------------------------
// Block reductions (256 threads)
// ---------------------------------------------------------------------------
__device__ __forceinline__ float block_reduce_sum(float v, float* sbuf) {
  int tid = threadIdx.x;
  sbuf[tid] = v; __syncthreads();
  #pragma unroll
  for (int s = 128; s > 0; s >>= 1) {
    if (tid < s) sbuf[tid] += sbuf[tid + s];
    __syncthreads();
  }
  float r = sbuf[0]; __syncthreads();
  return r;
}

__device__ __forceinline__ float block_reduce_max(float v, float* sbuf) {
  int tid = threadIdx.x;
  sbuf[tid] = v; __syncthreads();
  #pragma unroll
  for (int s = 128; s > 0; s >>= 1) {
    if (tid < s) sbuf[tid] = fmaxf(sbuf[tid], sbuf[tid + s]);
    __syncthreads();
  }
  float r = sbuf[0]; __syncthreads();
  return r;
}

// ---------------------------------------------------------------------------
// Weight convert + transpose: wt[n][k] = (bf16) w[k][n]
// ---------------------------------------------------------------------------
__global__ __launch_bounds__(256) void convert_wt_kernel(
    const float* __restrict__ w, bf16_t* __restrict__ wt, int K, int N) {
  int i = blockIdx.x * 256 + threadIdx.x;
  if (i >= K * N) return;
  int k = i / N, n = i - k * N;
  wt[(size_t)n * K + k] = (bf16_t)w[i];
}

__global__ __launch_bounds__(256) void concat3_kernel(
    float* __restrict__ dst, const float* __restrict__ a,
    const float* __restrict__ b, const float* __restrict__ c) {
  int i = blockIdx.x * 256 + threadIdx.x;
  if (i >= QKVN) return;
  dst[i] = (i < CH) ? a[i] : (i < 2 * CH) ? b[i - CH] : c[i - 2 * CH];
}

// ---------------------------------------------------------------------------
// Embedding gather + LayerNorm + per-token Gaussian noise
// ---------------------------------------------------------------------------
__global__ __launch_bounds__(256) void embed_kernel(
    const int* __restrict__ ids, const int* __restrict__ tts,
    const float* __restrict__ we, const float* __restrict__ pe,
    const float* __restrict__ te, const float* __restrict__ g,
    const float* __restrict__ bb, const float* __restrict__ imp,
    const float* __restrict__ noise,
    float* __restrict__ xf, bf16_t* __restrict__ xb) {
  __shared__ float sbuf[256];
  int tok = blockIdx.x;
  int s = tok % CS;
  int id = ids[tok];
  int tt = tts[tok];
  float e[3];
  float part = 0.f;
  #pragma unroll
  for (int j = 0; j < 3; ++j) {
    int f = threadIdx.x + j * 256;
    e[j] = we[(size_t)id * CH + f] + pe[(size_t)s * CH + f] + te[(size_t)tt * CH + f];
    part += e[j];
  }
  float mean = block_reduce_sum(part, sbuf) * (1.f / CH);
  float pv = 0.f;
  #pragma unroll
  for (int j = 0; j < 3; ++j) { float d = e[j] - mean; pv += d * d; }
  float var = block_reduce_sum(pv, sbuf) * (1.f / CH);
  float rinv = rsqrtf(var + 1e-12f);
  float sc = imp[tok];
  float sig = 1.f - sc;
  bool valid = (sc != -INFINITY);
  #pragma unroll
  for (int j = 0; j < 3; ++j) {
    int f = threadIdx.x + j * 256;
    float y = (e[j] - mean) * rinv * g[f] + bb[f];
    float out = valid ? (y + noise[(size_t)tok * CH + f] * 0.5f * sig * y) : y;
    size_t off = (size_t)tok * CH + f;
    xf[off] = out;
    xb[off] = (bf16_t)out;
  }
}

// ---------------------------------------------------------------------------
// Residual + LayerNorm: out = LN(resid + y) * g + b ; writes f32 and bf16
// ---------------------------------------------------------------------------
__global__ __launch_bounds__(256) void ln_residual_kernel(
    const float* __restrict__ resid, const float* __restrict__ y,
    const float* __restrict__ g, const float* __restrict__ bb,
    float* __restrict__ xf, bf16_t* __restrict__ xb) {
  __shared__ float sbuf[256];
  int row = blockIdx.x;
  float e[3];
  float part = 0.f;
  #pragma unroll
  for (int j = 0; j < 3; ++j) {
    size_t off = (size_t)row * CH + threadIdx.x + j * 256;
    e[j] = resid[off] + y[off];
    part += e[j];
  }
  float mean = block_reduce_sum(part, sbuf) * (1.f / CH);
  float pv = 0.f;
  #pragma unroll
  for (int j = 0; j < 3; ++j) { float d = e[j] - mean; pv += d * d; }
  float var = block_reduce_sum(pv, sbuf) * (1.f / CH);
  float rinv = rsqrtf(var + 1e-12f);
  #pragma unroll
  for (int j = 0; j < 3; ++j) {
    int f = threadIdx.x + j * 256;
    float out = (e[j] - mean) * rinv * g[f] + bb[f];
    size_t off = (size_t)row * CH + f;
    xf[off] = out;
    xb[off] = (bf16_t)out;
  }
}

// ---------------------------------------------------------------------------
// Main bf16 WMMA GEMM:  C[M][N] = A[M][K] * Wt[N][K]^T + bias  (+ GELU)
// Block = 256 threads = 8 waves. Block tile = 128 rows x 64 cols.
// Weight tile (64 rows x 64 K, 8KB) double-buffered in LDS via async copies.
// ---------------------------------------------------------------------------
__global__ __launch_bounds__(256) void gemm_bf16_kernel(
    const bf16_t* __restrict__ A, const bf16_t* __restrict__ Wt,
    const float* __restrict__ bias, float* __restrict__ Cf,
    bf16_t* __restrict__ Cb, int K, int lda, int ldw, int ldc, int gelu) {
  __shared__ bf16_t bsm[2][64 * 64];      // 2 x 8KB stage buffers
  unsigned tid = threadIdx.x;
  unsigned wave = tid >> 5, lane = tid & 31;
  int row0 = blockIdx.y * 128 + wave * 16;
  int col0 = blockIdx.x * 64;

  // Cooperative stage copy mapping: thread -> (row, 16-elem segment)
  int lrow = tid >> 2;                    // 0..63  (weight row within tile)
  int lseg = (tid & 3) * 16;              // 0,16,32,48 (elements)
  const bf16_t* wrow = Wt + (size_t)(col0 + lrow) * ldw + lseg;
  bf16_t* ldst0 = &bsm[0][lrow * 64 + lseg];
  bf16_t* ldst1 = &bsm[1][lrow * 64 + lseg];

  // Stage 0 in flight
  async_copy_b128(ldst0, wrow);
  async_copy_b128(ldst0 + 8, wrow + 8);

  v8f acc[4] = {};
  int nstage = K >> 6;
  for (int s = 0; s < nstage; ++s) {
    int kbase = s << 6;
    if (s + 1 < nstage) {                 // launch next stage copy
      const bf16_t* wnext = wrow + kbase + 64;
      bf16_t* ldst = (s & 1) ? ldst0 : ldst1;
      async_copy_b128(ldst, wnext);
      async_copy_b128(ldst + 8, wnext + 8);
      // prefetch next A stage for this wave's rows
      __builtin_prefetch(A + (size_t)(row0 + (lane & 15)) * lda + kbase + 64, 0, 0);
    }
    wait_async_le(s + 1 < nstage ? 2 : 0); // this stage's 2 copies are done
    __syncthreads();                       // stage visible to all waves
    const bf16_t* buf = &bsm[s & 1][0];
    #pragma unroll
    for (int kstep = 0; kstep < 2; ++kstep) {
      // hoist all fragment loads so they form one clause + single wait
      v16bf a = load_frag(A, row0, lda, kbase + kstep * 32, lane);
      v16bf b0 = lds_frag(buf, 0,  kstep, lane);
      v16bf b1 = lds_frag(buf, 16, kstep, lane);
      v16bf b2 = lds_frag(buf, 32, kstep, lane);
      v16bf b3 = lds_frag(buf, 48, kstep, lane);
      acc[0] = __builtin_amdgcn_wmma_f32_16x16x32_bf16(
          false, a, false, b0, (short)0, acc[0], false, false);
      acc[1] = __builtin_amdgcn_wmma_f32_16x16x32_bf16(
          false, a, false, b1, (short)0, acc[1], false, false);
      acc[2] = __builtin_amdgcn_wmma_f32_16x16x32_bf16(
          false, a, false, b2, (short)0, acc[2], false, false);
      acc[3] = __builtin_amdgcn_wmma_f32_16x16x32_bf16(
          false, a, false, b3, (short)0, acc[3], false, false);
    }
    __syncthreads();                       // all reads done before overwrite
  }

  int n_lo = lane & 15;
  int m_hi = (lane >> 4) * 8;
  #pragma unroll
  for (int t = 0; t < 4; ++t) {
    int n = col0 + t * 16 + n_lo;
    float bv = bias ? bias[n] : 0.f;
    #pragma unroll
    for (int r = 0; r < 8; ++r) {
      int m = row0 + m_hi + r;
      float v = acc[t][r] + bv;
      if (gelu) v = 0.5f * v * (1.f + erff(v * 0.70710678118f));
      size_t off = (size_t)m * ldc + n;
      if (Cf) Cf[off] = v;
      if (Cb) Cb[off] = (bf16_t)v;
    }
  }
}

// ---------------------------------------------------------------------------
// Attention raw scores: scores[bh][q][k] = Q[bh][q][:] . K[bh][k][:]
// Grid: (S/64, S/128, B*NH);  K = 64 -> direct global fragment loads.
// ---------------------------------------------------------------------------
__global__ __launch_bounds__(256) void attn_scores_kernel(
    const bf16_t* __restrict__ qkvb, float* __restrict__ scores) {
  unsigned tid = threadIdx.x;
  unsigned wave = tid >> 5, lane = tid & 31;
  int bh = blockIdx.z;
  int b = bh / CNH, h = bh - b * CNH;
  int row0 = blockIdx.y * 128 + wave * 16;
  int col0 = blockIdx.x * 64;
  const bf16_t* Q  = qkvb + (size_t)b * CS * QKVN + h * CDH;
  const bf16_t* Kp = qkvb + (size_t)b * CS * QKVN + CH + h * CDH;
  v8f acc[4] = {};
  #pragma unroll
  for (int kk = 0; kk < CDH; kk += 32) {
    v16bf a  = load_frag(Q, row0, QKVN, kk, lane);
    v16bf b0 = load_frag(Kp, col0 + 0,  QKVN, kk, lane);
    v16bf b1 = load_frag(Kp, col0 + 16, QKVN, kk, lane);
    v16bf b2 = load_frag(Kp, col0 + 32, QKVN, kk, lane);
    v16bf b3 = load_frag(Kp, col0 + 48, QKVN, kk, lane);
    acc[0] = __builtin_amdgcn_wmma_f32_16x16x32_bf16(
        false, a, false, b0, (short)0, acc[0], false, false);
    acc[1] = __builtin_amdgcn_wmma_f32_16x16x32_bf16(
        false, a, false, b1, (short)0, acc[1], false, false);
    acc[2] = __builtin_amdgcn_wmma_f32_16x16x32_bf16(
        false, a, false, b2, (short)0, acc[2], false, false);
    acc[3] = __builtin_amdgcn_wmma_f32_16x16x32_bf16(
        false, a, false, b3, (short)0, acc[3], false, false);
  }
  float* out = scores + (size_t)bh * CS * CS;
  int n_lo = lane & 15;
  int m_hi = (lane >> 4) * 8;
  #pragma unroll
  for (int t = 0; t < 4; ++t) {
    int n = col0 + t * 16 + n_lo;
    #pragma unroll
    for (int r = 0; r < 8; ++r)
      out[(size_t)(row0 + m_hi + r) * CS + n] = acc[t][r];
  }
}

// ---------------------------------------------------------------------------
// Softmax over keys with scale 1/sqrt(64) and attention-mask bias.
// ---------------------------------------------------------------------------
__global__ __launch_bounds__(256) void softmax_kernel(
    float* __restrict__ sc, const int* __restrict__ amask,
    bf16_t* __restrict__ pb) {
  __shared__ float sbuf[256];
  int r = blockIdx.x;          // (b*NH + h)*S + q
  int bh = r / CS;
  int b = bh / CNH;
  int k = threadIdx.x;
  size_t off = (size_t)r * CS + k;
  float v = sc[off] * 0.125f + (1.f - (float)amask[b * CS + k]) * -10000.f;
  float mx = block_reduce_max(v, sbuf);
  float e = __expf(v - mx);
  float sum = block_reduce_sum(e, sbuf);
  float p = e / sum;
  sc[off] = p;
  pb[off] = (bf16_t)p;
}

// ---------------------------------------------------------------------------
// V transpose per head: vT[bh][d][s] = V[b][s][h][d] (from fused QKV buffer)
// ---------------------------------------------------------------------------
__global__ __launch_bounds__(256) void vtrans_kernel(
    const bf16_t* __restrict__ qkvb, bf16_t* __restrict__ vT) {
  int i = blockIdx.x * 256 + threadIdx.x;   // over B*NH*DH*S (exact multiple)
  int s = i % CS;
  int r = i / CS;
  int d = r % CDH; r /= CDH;
  int h = r % CNH;
  int b = r / CNH;
  vT[i] = qkvb[(size_t)(b * CS + s) * QKVN + 2 * CH + h * CDH + d];
}

// ---------------------------------------------------------------------------
// Context: ctx[b][q][h][d] = sum_k probs[bh][q][k] * V[bh][k][d]
// A = probs bf16 [S][S]; B = vT [DH][S]. Grid: (1, S/128, B*NH)
// ---------------------------------------------------------------------------
__global__ __launch_bounds__(256) void attn_ctx_kernel(
    const bf16_t* __restrict__ probsb, const bf16_t* __restrict__ vT,
    bf16_t* __restrict__ ctxb) {
  unsigned tid = threadIdx.x;
  unsigned wave = tid >> 5, lane = tid & 31;
  int bh = blockIdx.z;
  int b = bh / CNH, h = bh - b * CNH;
  int row0 = blockIdx.y * 128 + wave * 16;
  int col0 = blockIdx.x * 64;
  const bf16_t* A  = probsb + (size_t)bh * CS * CS;
  const bf16_t* Bv = vT + (size_t)bh * CDH * CS;
  v8f acc[4] = {};
  for (int kk = 0; kk < CS; kk += 32) {
    v16bf a  = load_frag(A, row0, CS, kk, lane);
    v16bf b0 = load_frag(Bv, col0 + 0,  CS, kk, lane);
    v16bf b1 = load_frag(Bv, col0 + 16, CS, kk, lane);
    v16bf b2 = load_frag(Bv, col0 + 32, CS, kk, lane);
    v16bf b3 = load_frag(Bv, col0 + 48, CS, kk, lane);
    acc[0] = __builtin_amdgcn_wmma_f32_16x16x32_bf16(
        false, a, false, b0, (short)0, acc[0], false, false);
    acc[1] = __builtin_amdgcn_wmma_f32_16x16x32_bf16(
        false, a, false, b1, (short)0, acc[1], false, false);
    acc[2] = __builtin_amdgcn_wmma_f32_16x16x32_bf16(
        false, a, false, b2, (short)0, acc[2], false, false);
    acc[3] = __builtin_amdgcn_wmma_f32_16x16x32_bf16(
        false, a, false, b3, (short)0, acc[3], false, false);
  }
  int n_lo = lane & 15;
  int m_hi = (lane >> 4) * 8;
  #pragma unroll
  for (int t = 0; t < 4; ++t) {
    int n = col0 + t * 16 + n_lo;     // d within head
    #pragma unroll
    for (int r = 0; r < 8; ++r) {
      int m = row0 + m_hi + r;        // query index within (b)
      ctxb[(size_t)(b * CS + m) * CH + h * CDH + n] = (bf16_t)acc[t][r];
    }
  }
}

// ---------------------------------------------------------------------------
// Pooler: pooled[b][n] = tanh(x[b,0,:] . pool_w[:,n] + pool_b[n])  (tiny)
// ---------------------------------------------------------------------------
__global__ __launch_bounds__(256) void pooler_kernel(
    const float* __restrict__ x, const float* __restrict__ pw,
    const float* __restrict__ pb, float* __restrict__ out) {
  int i = blockIdx.x * 256 + threadIdx.x;
  if (i >= CB * CH) return;
  int b = i / CH, n = i - b * CH;
  const float* xr = x + (size_t)b * CS * CH;
  float acc = pb[n];
  for (int k = 0; k < CH; ++k) acc += xr[k] * pw[(size_t)k * CH + n];
  out[i] = tanhf(acc);
}

// ---------------------------------------------------------------------------
// Host orchestration
// ---------------------------------------------------------------------------
extern "C" void kernel_launch(void* const* d_in, const int* in_sizes, int n_in,
                              void* d_out, int out_size, void* d_ws,
                              size_t ws_size, hipStream_t stream) {
  (void)in_sizes; (void)n_in; (void)out_size; (void)ws_size;

  const int*   ids   = (const int*)d_in[0];
  const int*   amask = (const int*)d_in[1];
  const int*   tts   = (const int*)d_in[2];
  const float* imp   = (const float*)d_in[3];
  /* d_in[4] = rationale_mask (unused by the reference forward) */
  const float* noise = (const float*)d_in[5];
  const float* we    = (const float*)d_in[6];
  const float* pe    = (const float*)d_in[7];
  const float* te    = (const float*)d_in[8];
  const float* elng  = (const float*)d_in[9];
  const float* elnb  = (const float*)d_in[10];
  const float* poolw = (const float*)d_in[11];
  const float* poolb = (const float*)d_in[12];
  auto LP = [&](int l, int j) -> const float* {
    return (const float*)d_in[13 + l * 16 + j];
  };
  // j: 0 wq 1 bq 2 wk 3 bk 4 wv 5 bv 6 wo 7 bo 8 ln1g 9 ln1b
  //    10 wi 11 bi 12 wo2 13 bo2 14 ln2g 15 ln2b

  // ---- workspace carve ----
  char* wp = (char*)d_ws;
  auto alloc = [&](size_t bytes) -> void* {
    void* p = (void*)wp;
    wp += (bytes + 255) & ~(size_t)255;
    return p;
  };
  constexpr size_t WT_QKV = (size_t)QKVN * CH;
  constexpr size_t WT_WO  = (size_t)CH * CH;
  constexpr size_t WT_WI  = (size_t)CFF * CH;
  constexpr size_t WT_WO2 = (size_t)CH * CFF;
  constexpr size_t WT_LAYER = WT_QKV + WT_WO + WT_WI + WT_WO2;

  bf16_t* wt_all   = (bf16_t*)alloc(WT_LAYER * CL * sizeof(bf16_t));
  float*  qkvbias  = (float*)alloc((size_t)CL * QKVN * sizeof(float));
  float*  xf       = (float*)alloc((size_t)CM * CH * sizeof(float));
  bf16_t* xb       = (bf16_t*)alloc((size_t)CM * CH * sizeof(bf16_t));
  bf16_t* qkvb     = (bf16_t*)alloc((size_t)CM * QKVN * sizeof(bf16_t));
  bf16_t* vT       = (bf16_t*)alloc((size_t)CB * CNH * CDH * CS * sizeof(bf16_t));
  bf16_t* probsb   = (bf16_t*)alloc((size_t)CB * CNH * CS * CS * sizeof(bf16_t));
  bf16_t* hb       = (bf16_t*)alloc((size_t)CM * CFF * sizeof(bf16_t));
  float*  yf       = (float*)alloc((size_t)CM * CH * sizeof(float));
  bf16_t* ctxb     = (bf16_t*)alloc((size_t)CM * CH * sizeof(bf16_t));

  // ---- output layout: x [B,S,H] | pooled [B,H] | attns [L,B,NH,S,S] ----
  float* out_x      = (float*)d_out;
  float* out_pooled = out_x + (size_t)CM * CH;
  float* out_attn   = out_pooled + (size_t)CB * CH;
  constexpr size_t ATTN_LAYER = (size_t)CB * CNH * CS * CS;

  auto cvt = [&](const float* src, bf16_t* dst, int K, int N) {
    int total = K * N;
    convert_wt_kernel<<<(total + 255) / 256, 256, 0, stream>>>(src, dst, K, N);
  };

  // ---- convert all weights (bf16, transposed to [N][K]) ----
  for (int l = 0; l < CL; ++l) {
    bf16_t* base = wt_all + (size_t)l * WT_LAYER;
    cvt(LP(l, 0), base, CH, CH);                            // wq -> rows 0..767
    cvt(LP(l, 2), base + WT_WO, CH, CH);                    // wk -> rows 768..1535
    cvt(LP(l, 4), base + 2 * WT_WO, CH, CH);                // wv -> rows 1536..2303
    cvt(LP(l, 6), base + WT_QKV, CH, CH);                   // wo
    cvt(LP(l, 10), base + WT_QKV + WT_WO, CH, CFF);         // wi  [3072][768]
    cvt(LP(l, 12), base + WT_QKV + WT_WO + WT_WI, CFF, CH); // wo2 [768][3072]
    concat3_kernel<<<(QKVN + 255) / 256, 256, 0, stream>>>(
        qkvbias + (size_t)l * QKVN, LP(l, 1), LP(l, 3), LP(l, 5));
  }

  // ---- embeddings + LN + noise ----
  embed_kernel<<<CM, 256, 0, stream>>>(ids, tts, we, pe, te, elng, elnb,
                                       imp, noise, xf, xb);

  // ---- transformer layers ----
  for (int l = 0; l < CL; ++l) {
    bf16_t* base   = wt_all + (size_t)l * WT_LAYER;
    bf16_t* wt_qkv = base;
    bf16_t* wt_wo  = base + WT_QKV;
    bf16_t* wt_wi  = base + WT_QKV + WT_WO;
    bf16_t* wt_wo2 = base + WT_QKV + WT_WO + WT_WI;
    float*  attn_l = out_attn + (size_t)l * ATTN_LAYER;

    // fused QKV projection: [2048 x 768] * [2304 x 768]^T -> bf16 qkv
    gemm_bf16_kernel<<<dim3(QKVN / 64, CM / 128), 256, 0, stream>>>(
        xb, wt_qkv, qkvbias + (size_t)l * QKVN, nullptr, qkvb,
        CH, CH, CH, QKVN, 0);

    // scores = Q.K^T  (raw, into attns slot)
    attn_scores_kernel<<<dim3(CS / 64, CS / 128, CB * CNH), 256, 0, stream>>>(
        qkvb, attn_l);

    // softmax (scale + mask bias); probs stay in d_out, bf16 copy for ctx
    softmax_kernel<<<CB * CNH * CS, 256, 0, stream>>>(attn_l, amask, probsb);

    // V transpose per head
    vtrans_kernel<<<(CB * CNH * CDH * CS) / 256, 256, 0, stream>>>(qkvb, vT);

    // ctx = probs . V
    attn_ctx_kernel<<<dim3(1, CS / 128, CB * CNH), 256, 0, stream>>>(
        probsb, vT, ctxb);

    // attention output projection
    gemm_bf16_kernel<<<dim3(CH / 64, CM / 128), 256, 0, stream>>>(
        ctxb, wt_wo, LP(l, 7), yf, nullptr, CH, CH, CH, CH, 0);

    // x = LN(x + ctx@wo + bo)
    ln_residual_kernel<<<CM, 256, 0, stream>>>(xf, yf, LP(l, 8), LP(l, 9),
                                               xf, xb);

    // h = gelu(x@wi + bi)  (bf16 only)
    gemm_bf16_kernel<<<dim3(CFF / 64, CM / 128), 256, 0, stream>>>(
        xb, wt_wi, LP(l, 11), nullptr, hb, CH, CH, CH, CFF, 1);

    // y = h@wo2 + bo2
    gemm_bf16_kernel<<<dim3(CH / 64, CM / 128), 256, 0, stream>>>(
        hb, wt_wo2, LP(l, 13), yf, nullptr, CFF, CFF, CFF, CH, 0);

    // x = LN(x + y); final layer writes straight into d_out x-slot
    float* x_dst = (l == CL - 1) ? out_x : xf;
    ln_residual_kernel<<<CM, 256, 0, stream>>>(xf, yf, LP(l, 14), LP(l, 15),
                                               x_dst, xb);
  }

  // ---- pooler ----
  pooler_kernel<<<(CB * CH + 255) / 256, 256, 0, stream>>>(
      out_x, poolw, poolb, out_pooled);
}
